// _MambaBlock_24721831756732
// MI455X (gfx1250) — matrix operand order repaired and checked by reading, not compiled, not run on hardware
//
#include <hip/hip_runtime.h>
#include <hip/hip_bf16.h>
#include <cstdint>
#include <cstddef>

// ---------------------------------------------------------------------------
// Mamba-2 block for MI455X (gfx1250, wave32, WMMA).
// ---------------------------------------------------------------------------

typedef __attribute__((ext_vector_type(16))) __bf16        v16bf;
typedef __attribute__((ext_vector_type(8)))  float         v8f;
typedef __attribute__((ext_vector_type(8)))  unsigned short ushort8;

constexpr int BSZ   = 2;
constexpr int SEQ   = 4096;
constexpr int DM    = 768;
constexpr int DST   = 16;
constexpr int DIN   = 1536;          // EXPAND * DM
constexpr int NH    = 24;
constexpr int HD    = 64;
constexpr int CONVD = DIN + 2*DST;   // 1568
constexpr int DPROJ = 2*DIN + 2*DST + NH;  // 3128
constexpr int DPROJP= 3136;          // padded to multiple of 64 for WMMA tiles
constexpr int MROWS = BSZ*SEQ;       // 8192
constexpr float EPSV = 1e-5f;

__device__ __forceinline__ unsigned short f2bf(float f) {
  unsigned int u = __float_as_uint(f);
  u += 0x7FFFu + ((u >> 16) & 1u);   // round-to-nearest-even
  return (unsigned short)(u >> 16);
}

// ---------------------------------------------------------------------------
// fp32 -> bf16 conversion (weights)
// ---------------------------------------------------------------------------
__global__ void k_f32_to_bf16(const float* __restrict__ in,
                              unsigned short* __restrict__ out, int n) {
  int i = blockIdx.x * blockDim.x + threadIdx.x;
  if (i < n) out[i] = f2bf(in[i]);
}

// ---------------------------------------------------------------------------
// LayerNorm over DM=768, one block per token row, output bf16
// ---------------------------------------------------------------------------
__global__ void __launch_bounds__(256)
k_layernorm(const float* __restrict__ x, const float* __restrict__ w,
            const float* __restrict__ b, unsigned short* __restrict__ u) {
  const int r   = blockIdx.x;
  const int tid = threadIdx.x;
  const float* xr = x + (size_t)r * DM;

  float v[3], s = 0.f, s2 = 0.f;
#pragma unroll
  for (int i = 0; i < 3; ++i) {
    v[i] = xr[tid + i * 256];
    s  += v[i];
    s2 += v[i] * v[i];
  }
  __shared__ float rs[256], rs2[256];
  rs[tid] = s; rs2[tid] = s2;
  __syncthreads();
  for (int st = 128; st > 0; st >>= 1) {
    if (tid < st) { rs[tid] += rs[tid + st]; rs2[tid] += rs2[tid + st]; }
    __syncthreads();
  }
  const float mu  = rs[0] * (1.f / DM);
  const float var = rs2[0] * (1.f / DM) - mu * mu;
  const float inv = rsqrtf(var + EPSV);
#pragma unroll
  for (int i = 0; i < 3; ++i) {
    int c = tid + i * 256;
    u[(size_t)r * DM + c] = f2bf((v[i] - mu) * inv * w[c] + b[c]);
  }
}

// ---------------------------------------------------------------------------
// bf16 WMMA GEMM:  C[M x ldc] (fp32) = A[M x K] (bf16) * B[N x K]^T (bf16)
// Block = 256 threads = 8 waves; each wave owns a 32(M) x 64(N) tile:
// 2 A fragments x 4 B fragments -> 8 WMMA per 32-wide K step (8 f32
// accumulators = 64 VGPRs). Block tile = 256M x 64N.
// A fragment (16x32 bf16): lane<16 holds row (lane), K {0..15};
// lane>=16 same row, K {16..31}  -> two b128 loads per lane.
// B fragment (32x16 bf16): lane holds col (lane&15), K = (lane>>4)*16 .. +15
// contiguous -> two b128 loads per lane (weights are [N,K] row-major).
// ---------------------------------------------------------------------------
__global__ void __launch_bounds__(256)
k_gemm_wmma(const unsigned short* __restrict__ A,
            const unsigned short* __restrict__ Bw,
            float* __restrict__ C,
            const float* __restrict__ resid,
            int K, int Nclamp, int Nw, int ldc) {
  const int lane  = threadIdx.x & 31;
  const int wave  = threadIdx.x >> 5;
  const int mBase = blockIdx.x * 256 + wave * 32;
  const int nBase = blockIdx.y * 64;
  const int rlo   = lane & 15;
  const int kh    = lane >> 4;

  v8f acc[2][4];
#pragma unroll
  for (int g = 0; g < 2; ++g)
#pragma unroll
    for (int s = 0; s < 4; ++s) acc[g][s] = v8f{};

  const unsigned short* arow0 = A + (size_t)(mBase + rlo) * K;
  const unsigned short* arow1 = A + (size_t)(mBase + 16 + rlo) * K;
  int n0 = nBase +  0 + rlo; if (n0 >= Nclamp) n0 = Nclamp - 1;
  int n1 = nBase + 16 + rlo; if (n1 >= Nclamp) n1 = Nclamp - 1;
  int n2 = nBase + 32 + rlo; if (n2 >= Nclamp) n2 = Nclamp - 1;
  int n3 = nBase + 48 + rlo; if (n3 >= Nclamp) n3 = Nclamp - 1;
  const unsigned short* bp[4];
  bp[0] = Bw + (size_t)n0 * K + kh * 16;
  bp[1] = Bw + (size_t)n1 * K + kh * 16;
  bp[2] = Bw + (size_t)n2 * K + kh * 16;
  bp[3] = Bw + (size_t)n3 * K + kh * 16;

  for (int k0 = 0; k0 < K; k0 += 32) {
    // Prefetch the streamed A operand ahead (emits global_prefetch_b8);
    // B/weights live in the 192MB L2 after first touch.
    __builtin_prefetch(arow0 + k0 + 256, 0, 1);
    __builtin_prefetch(arow1 + k0 + 256, 0, 1);

    union { v16bf v; ushort8 h[2]; } af0, af1, bf[4];
    af0.h[0] = *(const ushort8*)(arow0 + k0 + kh * 16);
    af0.h[1] = *(const ushort8*)(arow0 + k0 + kh * 16 + 8);
    af1.h[0] = *(const ushort8*)(arow1 + k0 + kh * 16);
    af1.h[1] = *(const ushort8*)(arow1 + k0 + kh * 16 + 8);
#pragma unroll
    for (int s = 0; s < 4; ++s) {
      bf[s].h[0] = *(const ushort8*)(bp[s] + k0);
      bf[s].h[1] = *(const ushort8*)(bp[s] + k0 + 8);
    }
#pragma unroll
    for (int s = 0; s < 4; ++s) {
      acc[0][s] = __builtin_amdgcn_wmma_f32_16x16x32_bf16(
          false, af0.v, false, bf[s].v, (short)0, acc[0][s], false, false);
      acc[1][s] = __builtin_amdgcn_wmma_f32_16x16x32_bf16(
          false, af1.v, false, bf[s].v, (short)0, acc[1][s], false, false);
    }
  }

  // C/D layout: VGPR i: lanes 0-15 -> M=i, lanes 16-31 -> M=i+8; N = lane&15
  const int cc[4] = { nBase + 0 + rlo, nBase + 16 + rlo,
                      nBase + 32 + rlo, nBase + 48 + rlo };
#pragma unroll
  for (int g = 0; g < 2; ++g) {
#pragma unroll
    for (int i = 0; i < 8; ++i) {
      const int r = mBase + g * 16 + kh * 8 + i;
      float* crow = C + (size_t)r * ldc;
      const float* rrow = resid ? (resid + (size_t)r * Nw) : nullptr;
#pragma unroll
      for (int s = 0; s < 4; ++s) {
        if (cc[s] < Nw)
          crow[cc[s]] = acc[g][s][i] + (rrow ? rrow[cc[s]] : 0.f);
      }
    }
  }
}

// ---------------------------------------------------------------------------
// Causal depthwise conv (width 4) + SiLU over the xBC slice of zxbcdt
// ---------------------------------------------------------------------------
__global__ void k_conv_silu(const float* __restrict__ zx,
                            const float* __restrict__ cw,
                            const float* __restrict__ cb,
                            float* __restrict__ out) {
  int idx = blockIdx.x * 256 + threadIdx.x;
  if (idx >= BSZ * SEQ * CONVD) return;
  const int c = idx % CONVD;
  const int l = (idx / CONVD) % SEQ;
  const int b = idx / (CONVD * SEQ);
  float acc = cb[c];
#pragma unroll
  for (int i = 0; i < 4; ++i) {
    const int li = l - 3 + i;
    if (li >= 0)
      acc += zx[((size_t)(b * SEQ + li)) * DPROJP + DIN + c] * cw[c * 4 + i];
  }
  out[idx] = acc / (1.f + __expf(-acc));   // silu
}

// ---------------------------------------------------------------------------
// dt = softplus(raw + dt_bias), dA = exp(-exp(A_log) * dt)
// ---------------------------------------------------------------------------
__global__ void k_dt(const float* __restrict__ zx,
                     const float* __restrict__ dt_bias,
                     const float* __restrict__ A_log,
                     float* __restrict__ dt, float* __restrict__ dA) {
  int idx = blockIdx.x * 256 + threadIdx.x;
  if (idx >= MROWS * NH) return;
  const int h = idx % NH;
  const int row = idx / NH;
  const float raw = zx[(size_t)row * DPROJP + DIN + CONVD + h] + dt_bias[h];
  const float d = raw > 20.f ? raw : log1pf(__expf(raw));
  dt[idx] = d;
  dA[idx] = __expf(-__expf(A_log[h]) * d);
}

// ---------------------------------------------------------------------------
// Sequential SSM scan. One block per (batch, head). 256 threads hold the
// 64x16 state: thread t owns p = t>>2, states n = (t&3)*4 .. +3.
// Chunks of 64 timesteps staged through LDS; y reduced over state dim with
// wave32 shfl_xor within lane quads.
// ---------------------------------------------------------------------------
__global__ void __launch_bounds__(256)
k_scan(const float* __restrict__ conv, const float* __restrict__ dt,
       const float* __restrict__ dA, const float* __restrict__ Dp,
       float* __restrict__ y) {
  const int b   = blockIdx.x / NH;
  const int h   = blockIdx.x % NH;
  const int tid = threadIdx.x;
  const int p   = tid >> 2;
  const int n4  = tid & 3;
  const float Dh = Dp[h];

  __shared__ float sx[64][64];
  __shared__ float sB[64][16];
  __shared__ float sC[64][16];
  __shared__ float sdt[64], sdA[64];
  __shared__ float sy[64][64];

  float hs[4] = {0.f, 0.f, 0.f, 0.f};

  for (int l0 = 0; l0 < SEQ; l0 += 64) {
#pragma unroll
    for (int i = 0; i < 16; ++i) {
      const int idx = tid + i * 256;
      const int t = idx >> 6, pp = idx & 63;
      sx[t][pp] = conv[((size_t)(b * SEQ + l0 + t)) * CONVD + h * HD + pp];
    }
#pragma unroll
    for (int i = 0; i < 4; ++i) {
      const int idx = tid + i * 256;
      const int t = idx >> 4, nn = idx & 15;
      sB[t][nn] = conv[((size_t)(b * SEQ + l0 + t)) * CONVD + DIN + nn];
      sC[t][nn] = conv[((size_t)(b * SEQ + l0 + t)) * CONVD + DIN + DST + nn];
    }
    if (tid < 64) {
      const int gi = (b * SEQ + l0 + tid) * NH + h;
      sdt[tid] = dt[gi];
      sdA[tid] = dA[gi];
    }
    __syncthreads();

    for (int t = 0; t < 64; ++t) {
      const float a  = sdA[t];
      const float xp = sx[t][p];
      const float sc = sdt[t] * xp;
      float yp = 0.f;
#pragma unroll
      for (int j = 0; j < 4; ++j) {
        const int n = n4 * 4 + j;
        hs[j] = a * hs[j] + sc * sB[t][n];
        yp   += hs[j] * sC[t][n];
      }
      yp += __shfl_xor(yp, 1, 32);
      yp += __shfl_xor(yp, 2, 32);
      if (n4 == 0) sy[t][p] = yp + Dh * xp;
    }
    __syncthreads();

#pragma unroll
    for (int i = 0; i < 16; ++i) {
      const int idx = tid + i * 256;
      const int t = idx >> 6, pp = idx & 63;
      y[((size_t)(b * SEQ + l0 + t)) * DIN + h * HD + pp] = sy[t][pp];
    }
    __syncthreads();
  }
}

// ---------------------------------------------------------------------------
// g = y * silu(z); RMS-norm over DIN with gate_norm_w; output bf16
// ---------------------------------------------------------------------------
__global__ void __launch_bounds__(256)
k_gate(const float* __restrict__ y, const float* __restrict__ zx,
       const float* __restrict__ gw, unsigned short* __restrict__ g) {
  const int r = blockIdx.x, tid = threadIdx.x;
  const float* yr = y + (size_t)r * DIN;
  const float* zr = zx + (size_t)r * DPROJP;

  float gv[6], s = 0.f;
#pragma unroll
  for (int i = 0; i < 6; ++i) {
    const int c = tid + i * 256;
    const float zv = zr[c];
    const float gg = yr[c] * (zv / (1.f + __expf(-zv)));
    gv[i] = gg;
    s += gg * gg;
  }
  __shared__ float red[256];
  red[tid] = s;
  __syncthreads();
  for (int st = 128; st > 0; st >>= 1) {
    if (tid < st) red[tid] += red[tid + st];
    __syncthreads();
  }
  const float inv = rsqrtf(red[0] * (1.f / DIN) + EPSV);
#pragma unroll
  for (int i = 0; i < 6; ++i) {
    const int c = tid + i * 256;
    g[(size_t)r * DIN + c] = f2bf(gv[i] * inv * gw[c]);
  }
}

// ---------------------------------------------------------------------------
extern "C" void kernel_launch(void* const* d_in, const int* in_sizes, int n_in,
                              void* d_out, int out_size, void* d_ws, size_t ws_size,
                              hipStream_t stream) {
  (void)in_sizes; (void)n_in; (void)out_size; (void)ws_size;
  const float* x    = (const float*)d_in[0];
  const float* lnw  = (const float*)d_in[1];
  const float* lnb  = (const float*)d_in[2];
  const float* Win  = (const float*)d_in[3];
  const float* cw   = (const float*)d_in[4];
  const float* cb   = (const float*)d_in[5];
  const float* dtb  = (const float*)d_in[6];
  const float* Alg  = (const float*)d_in[7];
  const float* Dp   = (const float*)d_in[8];
  const float* gnw  = (const float*)d_in[9];
  const float* Wout = (const float*)d_in[10];
  float* out = (float*)d_out;

  char* ws = (char*)d_ws;
  size_t off = 0;
  auto alloc = [&](size_t bytes) -> void* {
    off = (off + 255) & ~(size_t)255;
    void* p = ws + off;
    off += bytes;
    return p;
  };
  unsigned short* u_bf    = (unsigned short*)alloc((size_t)MROWS * DM * 2);
  unsigned short* Win_bf  = (unsigned short*)alloc((size_t)DPROJ * DM * 2);
  unsigned short* Wout_bf = (unsigned short*)alloc((size_t)DM * DIN * 2);
  float*          zx      = (float*)alloc((size_t)MROWS * DPROJP * 4);
  float*          cv      = (float*)alloc((size_t)MROWS * CONVD * 4);
  float*          dtv     = (float*)alloc((size_t)MROWS * NH * 4);
  float*          dAv     = (float*)alloc((size_t)MROWS * NH * 4);
  float*          yv      = (float*)alloc((size_t)MROWS * DIN * 4);
  unsigned short* g_bf    = (unsigned short*)alloc((size_t)MROWS * DIN * 2);

  const int nWin = DPROJ * DM;
  k_f32_to_bf16<<<(nWin + 255) / 256, 256, 0, stream>>>(Win, Win_bf, nWin);
  const int nWout = DM * DIN;
  k_f32_to_bf16<<<(nWout + 255) / 256, 256, 0, stream>>>(Wout, Wout_bf, nWout);

  k_layernorm<<<MROWS, 256, 0, stream>>>(x, lnw, lnb, u_bf);

  // GEMM1: zx[8192 x 3136] = u @ W_in^T  (K=768, valid N = 3128)
  k_gemm_wmma<<<dim3(MROWS / 256, DPROJP / 64), 256, 0, stream>>>(
      u_bf, Win_bf, zx, nullptr, DM, DPROJ, DPROJP, DPROJP);

  const int nConv = MROWS * CONVD;
  k_conv_silu<<<(nConv + 255) / 256, 256, 0, stream>>>(zx, cw, cb, cv);

  const int nDt = MROWS * NH;
  k_dt<<<(nDt + 255) / 256, 256, 0, stream>>>(zx, dtb, Alg, dtv, dAv);

  k_scan<<<BSZ * NH, 256, 0, stream>>>(cv, dtv, dAv, Dp, yv);

  k_gate<<<MROWS, 256, 0, stream>>>(yv, zx, gnw, g_bf);

  // GEMM2: out[8192 x 768] = g @ W_out^T + x  (K=1536, fused residual)
  k_gemm_wmma<<<dim3(MROWS / 256, DM / 64), 256, 0, stream>>>(
      g_bf, Wout_bf, out, x, DIN, DM, DM, DM);
}
